// DecoderRNN_15109694948129
// MI455X (gfx1250) — matrix operand order repaired and checked by reading, not compiled, not run on hardware
//
#include <hip/hip_runtime.h>
#include <math.h>

// ---------------------------------------------------------------------------
// Problem constants (fixed by the reference harness)
// ---------------------------------------------------------------------------
#define BSZ   128          // batch
#define HS    512          // hidden
#define ES    256          // embedding
#define VS    1000         // vocab
#define TS    256          // seq len
#define LS    512          // encoder len
#define KIH   768          // E + H
#define N3H   1536         // 3*H
#define VPAD  1024         // vocab padded to multiple of 64 (N-supertile)

// ---------------------------------------------------------------------------
// Types for WMMA (CDNA5 / gfx1250, wave32)
// ---------------------------------------------------------------------------
typedef __bf16          bf16x16 __attribute__((ext_vector_type(16)));
typedef float           f32x8   __attribute__((ext_vector_type(8)));
typedef unsigned short  u16x8   __attribute__((ext_vector_type(8)));

union BfVec { bf16x16 v; u16x8 u[2]; };

__device__ __forceinline__ unsigned short f2bf(float f) {
    unsigned u = __float_as_uint(f);
    unsigned r = (u + 0x7FFFu + ((u >> 16) & 1u)) >> 16;   // round-nearest-even
    return (unsigned short)r;
}

// ---------------------------------------------------------------------------
// Prep kernels: fp32 -> bf16 weight conversion, hidden-state init
// ---------------------------------------------------------------------------
__global__ void __launch_bounds__(256)
cvt_bf16_kernel(const float* __restrict__ src, unsigned short* __restrict__ dst, int n) {
    int i = blockIdx.x * 256 + threadIdx.x;
    if (i < n) dst[i] = f2bf(src[i]);
}

// W_lin [1000,512] -> padded bf16 [1024,512] (zero rows 1000..1023)
__global__ void __launch_bounds__(256)
cvt_wlin_kernel(const float* __restrict__ src, unsigned short* __restrict__ dst) {
    int i = blockIdx.x * 256 + threadIdx.x;        // < 1024*512
    int row = i >> 9;
    float v = (row < VS) ? src[i] : 0.f;
    dst[i] = f2bf(v);
}

__global__ void __launch_bounds__(256)
init_h_kernel(const float* __restrict__ hidden, float* __restrict__ h,
              unsigned short* __restrict__ hbf) {
    int i = blockIdx.x * 256 + threadIdx.x;        // < BSZ*HS
    float v = hidden[i];
    h[i] = v;
    hbf[i] = f2bf(v);
}

// ---------------------------------------------------------------------------
// Attention + input build. One block per batch row. fp32, L2-resident streams.
// Produces x_bf16[b, 0:256]   = emb_table[formulas[b,t]]
//          x_bf16[b, 256:768] = softmax(enc[b] @ h[b]) @ enc[b]
// ---------------------------------------------------------------------------
__global__ void __launch_bounds__(256)
attention_x_kernel(const float* __restrict__ enc,
                   const float* __restrict__ h,
                   const float* __restrict__ emb_table,
                   const int*   __restrict__ formulas,
                   unsigned short* __restrict__ x_bf,
                   int t) {
    __shared__ __align__(16) float sh_h[HS];
    __shared__ __align__(16) float sc[LS];
    __shared__ float red[256];

    const int tid = threadIdx.x;
    const int b   = blockIdx.x;
    const float* __restrict__ encb = enc + (size_t)b * LS * HS;

    sh_h[tid]       = h[b * HS + tid];
    sh_h[tid + 256] = h[b * HS + tid + 256];
    __syncthreads();

    // scores[l] = enc[b,l,:] . h[b,:]
    for (int l = tid; l < LS; l += 256) {
        const float4* row = (const float4*)(encb + l * HS);
        const float4* hv4 = (const float4*)sh_h;
        float acc = 0.f;
        #pragma unroll 4
        for (int k = 0; k < HS / 4; ++k) {
            float4 e = row[k], hv = hv4[k];
            acc += e.x * hv.x + e.y * hv.y + e.z * hv.z + e.w * hv.w;
        }
        sc[l] = acc;
    }
    __syncthreads();

    // softmax over L=512 (block reduction)
    float m = fmaxf(sc[tid], sc[tid + 256]);
    red[tid] = m; __syncthreads();
    for (int s = 128; s > 0; s >>= 1) {
        if (tid < s) red[tid] = fmaxf(red[tid], red[tid + s]);
        __syncthreads();
    }
    const float mx = red[0];
    __syncthreads();
    float e0 = __expf(sc[tid] - mx), e1 = __expf(sc[tid + 256] - mx);
    sc[tid] = e0; sc[tid + 256] = e1;
    red[tid] = e0 + e1; __syncthreads();
    for (int s = 128; s > 0; s >>= 1) {
        if (tid < s) red[tid] += red[tid + s];
        __syncthreads();
    }
    const float inv = 1.f / red[0];
    __syncthreads();

    // context[hh] = sum_l coef[l] * enc[b,l,hh]   (coalesced across tid)
    float c0 = 0.f, c1 = 0.f;
    for (int l = 0; l < LS; ++l) {
        const float w = sc[l];
        c0 += w * encb[l * HS + tid];
        c1 += w * encb[l * HS + tid + 256];
    }

    unsigned short* __restrict__ xr = x_bf + b * KIH;
    xr[ES + tid]       = f2bf(c0 * inv);
    xr[ES + 256 + tid] = f2bf(c1 * inv);

    // embedding part (E=256, one element per thread)
    const int tok = formulas[b * TS + t];
    xr[tid] = f2bf(emb_table[tok * ES + tid]);
}

// ---------------------------------------------------------------------------
// Register-blocked bf16 WMMA GEMM: each wave owns a 32(M) x 64(N) output tile
// (MB=2 A-fragments x NB=4 B-fragments -> 8 v_wmma per K-step of 32).
// C[M,N] = A[M,K] @ W[N,K]^T, fp32 accumulate.
//   A-operand: lane l / l+16 hold row l, K halves {k..k+7,k+16..k+23} /
//              {k+8..k+15,k+24..k+31} (two 16B loads).
//   B-operand: lane n / n+16 hold W row n, same K halves as A
//              (two 16B loads) -- no transpose needed since W is [N,K].
// VOCAB=true additionally adds bias, masks col<VS, and remaps rows
// (t*128+b) -> out[(b*TS+t)*VS + col].
// ---------------------------------------------------------------------------
template<bool VOCAB>
__global__ void __launch_bounds__(128)
gemm_bf16_wmma_blk(const unsigned short* __restrict__ A,
                   const unsigned short* __restrict__ W,
                   float* __restrict__ C,
                   const float* __restrict__ bias,
                   int nsuper, int K, int N) {
    const int lane = threadIdx.x & 31;
    const int wave = blockIdx.x * 4 + (threadIdx.x >> 5);
    const int mt = wave / nsuper, nt = wave % nsuper;
    const int half = lane >> 4, l15 = lane & 15;

    const unsigned short* __restrict__ a0 = A + (size_t)(mt * 32 + l15) * K + half * 8;
    const unsigned short* __restrict__ a1 = a0 + (size_t)16 * K;
    const unsigned short* __restrict__ w0 = W + (size_t)(nt * 64 + l15) * K + half * 8;
    const size_t wstep = (size_t)16 * K;

    const f32x8 zero = {0.f, 0.f, 0.f, 0.f, 0.f, 0.f, 0.f, 0.f};
    f32x8 acc[2][4];
    #pragma unroll
    for (int mi = 0; mi < 2; ++mi)
        #pragma unroll
        for (int ni = 0; ni < 4; ++ni) acc[mi][ni] = zero;

    for (int k = 0; k < K; k += 32) {
        BfVec av[2], wv[4];
        av[0].u[0] = *(const u16x8*)(a0 + k);
        av[0].u[1] = *(const u16x8*)(a0 + k + 16);
        av[1].u[0] = *(const u16x8*)(a1 + k);
        av[1].u[1] = *(const u16x8*)(a1 + k + 16);
        #pragma unroll
        for (int ni = 0; ni < 4; ++ni) {
            wv[ni].u[0] = *(const u16x8*)(w0 + (size_t)ni * wstep + k);
            wv[ni].u[1] = *(const u16x8*)(w0 + (size_t)ni * wstep + k + 16);
        }
        #pragma unroll
        for (int mi = 0; mi < 2; ++mi)
            #pragma unroll
            for (int ni = 0; ni < 4; ++ni)
                acc[mi][ni] = __builtin_amdgcn_wmma_f32_16x16x32_bf16(
                    false, av[mi].v, false, wv[ni].v, (short)0, acc[mi][ni],
                    false, false);
    }

    // D layout: VGPR r -> row (+half*8 + r), col (+l15)
    if (!VOCAB) {
        #pragma unroll
        for (int mi = 0; mi < 2; ++mi) {
            float* __restrict__ c =
                C + (size_t)(mt * 32 + mi * 16 + half * 8) * N + nt * 64 + l15;
            #pragma unroll
            for (int ni = 0; ni < 4; ++ni)
                #pragma unroll
                for (int r = 0; r < 8; ++r)
                    c[(size_t)r * N + ni * 16] = acc[mi][ni][r];
        }
    } else {
        #pragma unroll
        for (int ni = 0; ni < 4; ++ni) {
            const int col = nt * 64 + ni * 16 + l15;
            if (col < VS) {
                const float bv = bias[col];
                #pragma unroll
                for (int mi = 0; mi < 2; ++mi)
                    #pragma unroll
                    for (int r = 0; r < 8; ++r) {
                        const int gr = mt * 32 + mi * 16 + half * 8 + r; // t*128+b
                        const int tt = gr >> 7, bb = gr & 127;
                        C[((size_t)bb * TS + tt) * VS + col] = acc[mi][ni][r] + bv;
                    }
            }
        }
    }
}

// ---------------------------------------------------------------------------
// GRU gates (PyTorch order r,z,n): h' = (1-z)*tanh(i_n + r*h_n) + z*h
// ---------------------------------------------------------------------------
__global__ void __launch_bounds__(256)
gru_gate_kernel(const float* __restrict__ gi, const float* __restrict__ gh,
                const float* __restrict__ b_ih, const float* __restrict__ b_hh,
                float* __restrict__ h, unsigned short* __restrict__ hbf,
                unsigned short* __restrict__ outs_bf, int t) {
    const int idx = blockIdx.x * 256 + threadIdx.x;   // < BSZ*HS
    const int b = idx >> 9, j = idx & (HS - 1);
    const int g = b * N3H + j;

    const float ir  = gi[g]           + b_ih[j];
    const float iz  = gi[g + HS]      + b_ih[j + HS];
    const float in_ = gi[g + 2 * HS]  + b_ih[j + 2 * HS];
    const float hr  = gh[g]           + b_hh[j];
    const float hz  = gh[g + HS]      + b_hh[j + HS];
    const float hn  = gh[g + 2 * HS]  + b_hh[j + 2 * HS];

    const float r  = 1.f / (1.f + __expf(-(ir + hr)));
    const float z  = 1.f / (1.f + __expf(-(iz + hz)));
    const float nn = tanhf(in_ + r * hn);

    const float hold = h[idx];
    const float hnew = (1.f - z) * nn + z * hold;
    h[idx]   = hnew;
    hbf[idx] = f2bf(hnew);
    outs_bf[((size_t)t * BSZ + b) * HS + j] = f2bf(hnew);
}

// ---------------------------------------------------------------------------
// Host launcher
// ---------------------------------------------------------------------------
extern "C" void kernel_launch(void* const* d_in, const int* in_sizes, int n_in,
                              void* d_out, int out_size, void* d_ws, size_t ws_size,
                              hipStream_t stream) {
    const float* hidden   = (const float*)d_in[0];
    /* d_in[1] = n (scalar, fixed = 128) */
    const int*   formulas = (const int*)d_in[2];
    const float* enc      = (const float*)d_in[3];
    const float* emb      = (const float*)d_in[4];
    const float* W_ih     = (const float*)d_in[5];
    const float* W_hh     = (const float*)d_in[6];
    const float* b_ih     = (const float*)d_in[7];
    const float* b_hh     = (const float*)d_in[8];
    const float* W_lin    = (const float*)d_in[9];
    const float* b_lin    = (const float*)d_in[10];
    float* out = (float*)d_out;

    // Workspace carve-up (256B aligned)
    char* ws = (char*)d_ws;
    size_t off = 0;
    auto take = [&](size_t bytes) -> char* {
        char* p = ws + off;
        off = (off + bytes + 255) & ~(size_t)255;
        return p;
    };
    unsigned short* W_ih_bf  = (unsigned short*)take((size_t)N3H * KIH * 2);     // 2.36 MB
    unsigned short* W_hh_bf  = (unsigned short*)take((size_t)N3H * HS * 2);      // 1.57 MB
    unsigned short* W_lin_bf = (unsigned short*)take((size_t)VPAD * HS * 2);     // 1.05 MB
    unsigned short* x_bf     = (unsigned short*)take((size_t)BSZ * KIH * 2);     // 0.20 MB
    float*          h_f32    = (float*)take((size_t)BSZ * HS * 4);               // 0.26 MB
    unsigned short* h_bf     = (unsigned short*)take((size_t)BSZ * HS * 2);      // 0.13 MB
    float*          gi       = (float*)take((size_t)BSZ * N3H * 4);              // 0.79 MB
    float*          gh       = (float*)take((size_t)BSZ * N3H * 4);              // 0.79 MB
    unsigned short* outs_bf  = (unsigned short*)take((size_t)TS * BSZ * HS * 2); // 33.6 MB

    // One-time preps
    cvt_bf16_kernel<<<(N3H * KIH + 255) / 256, 256, 0, stream>>>(W_ih, W_ih_bf, N3H * KIH);
    cvt_bf16_kernel<<<(N3H * HS + 255) / 256, 256, 0, stream>>>(W_hh, W_hh_bf, N3H * HS);
    cvt_wlin_kernel<<<(VPAD * HS) / 256, 256, 0, stream>>>(W_lin, W_lin_bf);
    init_h_kernel<<<(BSZ * HS) / 256, 256, 0, stream>>>(hidden, h_f32, h_bf);

    // Sequential scan over T steps
    const int gate_nsuper = N3H / 64;                         // 24 N-supertiles
    const int gate_waves  = (BSZ / 32) * gate_nsuper;         // 4*24 = 96 waves
    for (int t = 0; t < TS; ++t) {
        attention_x_kernel<<<BSZ, 256, 0, stream>>>(enc, h_f32, emb, formulas, x_bf, t);
        // gi = x @ W_ih^T : M=128, N=1536, K=768
        gemm_bf16_wmma_blk<false><<<gate_waves / 4, 128, 0, stream>>>(
            x_bf, W_ih_bf, gi, nullptr, gate_nsuper, KIH, N3H);
        // gh = h @ W_hh^T : M=128, N=1536, K=512
        gemm_bf16_wmma_blk<false><<<gate_waves / 4, 128, 0, stream>>>(
            h_bf, W_hh_bf, gh, nullptr, gate_nsuper, HS, N3H);
        gru_gate_kernel<<<(BSZ * HS) / 256, 256, 0, stream>>>(gi, gh, b_ih, b_hh,
                                                              h_f32, h_bf, outs_bf, t);
    }

    // Final vocab projection: M=32768 (1024 M-supertiles), N=1024 (16 N-supertiles)
    const int voc_nsuper = VPAD / 64;                          // 16
    const int voc_waves  = (TS * BSZ / 32) * voc_nsuper;       // 1024*16 = 16384
    gemm_bf16_wmma_blk<true><<<voc_waves / 4, 128, 0, stream>>>(
        outs_bf, W_lin_bf, out, b_lin, voc_nsuper, HS, VPAD);
}
